// CausalSelfAttention_25417616458111
// MI455X (gfx1250) — matrix operand order repaired and checked by reading, not compiled, not run on hardware
//
#include <hip/hip_runtime.h>
#include <hip/hip_bf16.h>

typedef _Float16 f16_t;
typedef __attribute__((ext_vector_type(8)))  _Float16 v8h;
typedef __attribute__((ext_vector_type(16))) _Float16 v16h;
typedef __attribute__((ext_vector_type(8)))  float    v8f;
typedef __attribute__((ext_vector_type(4)))  unsigned int u32x4;
typedef __attribute__((ext_vector_type(8)))  int i32x8;
typedef __attribute__((ext_vector_type(4)))  int i32x4;

#define NEMBD 1024
#define NHEAD 16
#define HDIM  64
#define TSEQ  2048
#define BATCH 4
#define ROWS  (BATCH * TSEQ)   // 8192
#define N3C   (3 * NEMBD)      // 3072

// ---- WMMA helpers -----------------------------------------------------------

__device__ __forceinline__ v8f wmma_f16(v16h a, v16h b, v8f c) {
  // v_wmma_f32_16x16x32_f16: D = A(16x32) * B(32x16) + C(16x16)
  return __builtin_amdgcn_wmma_f32_16x16x32_f16(false, a, false, b, (short)0, c,
                                                false, false);
}

// A-fragment (16x32, 16-bit): lane r=lane&15 is row M; half=lane>>4 selects
// K = half*16 + e for e = 0..15 (contiguous).
// p = &src[row*ld + kbase + half*16]
__device__ __forceinline__ v16h ld_a32(const f16_t* p) {
  v8h lo = *(const v8h*)p;
  v8h hi = *(const v8h*)(p + 8);
  v16h r;
#pragma unroll
  for (int i = 0; i < 8; ++i) { r[i] = lo[i]; r[i + 8] = hi[i]; }
  return r;
}

// B-fragment (32x16, 16-bit): lane r=lane&15 is column N; half=lane>>4 selects
// K = half*16 + e for e = 0..15 (contiguous).  Source must be N-major.
// p = &src[n*ld + kbase + half*16]
__device__ __forceinline__ v16h ld_b32(const f16_t* p) {
  v8h lo = *(const v8h*)p;
  v8h hi = *(const v8h*)(p + 8);
  v16h r;
#pragma unroll
  for (int i = 0; i < 8; ++i) { r[i] = lo[i]; r[i + 8] = hi[i]; }
  return r;
}

__device__ __forceinline__ v8f vzero8() {
  v8f z;
#pragma unroll
  for (int i = 0; i < 8; ++i) z[i] = 0.0f;
  return z;
}

// ---- Tensor Data Mover: 2-D tile (f16) global -> LDS -----------------------
// D# per CDNA5 ISA ch.8: group0 = {count=1, lds_addr, global_addr(57b), type=2}
// group1 = {data_size=2B, tensor_dim0/1, tile_dim0/1, tensor_dim0_stride}
// remaining groups zero (<=2D tensor).  Tracked by TENSORcnt.

__device__ __forceinline__ void tdm_load_2d(unsigned int lds_addr,
                                            const void* gptr,
                                            unsigned int tensor_d0,
                                            unsigned int tensor_d1,
                                            unsigned int tile_d0,
                                            unsigned int tile_d1,
                                            unsigned int stride0) {
  unsigned long long ga = (unsigned long long)(uintptr_t)gptr;
  u32x4 g0;
  g0[0] = 1u;                                    // count=1, user descriptor
  g0[1] = lds_addr;                              // LDS byte address
  g0[2] = (unsigned int)ga;                      // global addr [31:0]
  g0[3] = (unsigned int)(ga >> 32) | 0x80000000u;// addr[56:32] | type=2<<30
  i32x8 g1;
  g1[0] = (int)0x00010000u;                      // data_size = 1 (2 bytes)
  g1[1] = (int)(tensor_d0 << 16);                // tensor_dim0[15:0] @63:48
  g1[2] = (int)((tensor_d0 >> 16) | (tensor_d1 << 16));
  g1[3] = (int)((tensor_d1 >> 16) | (tile_d0 << 16));
  g1[4] = (int)(tile_d1 & 0xffffu);              // tile_dim1; tile_dim2=0
  g1[5] = (int)stride0;                          // tensor_dim0_stride[31:0]
  g1[6] = 0;
  g1[7] = 0;
  i32x4 z4;
  z4[0] = 0; z4[1] = 0; z4[2] = 0; z4[3] = 0;
  i32x8 z8;
#pragma unroll
  for (int i = 0; i < 8; ++i) z8[i] = 0;
  __builtin_amdgcn_tensor_load_to_lds(g0, g1, z4, z4, z8, 0);
}

// ---- Prep kernels -----------------------------------------------------------

__global__ __launch_bounds__(256) void k_f32_to_f16(const float* __restrict__ s,
                                                    f16_t* __restrict__ d, int n) {
  int i = blockIdx.x * 256 + threadIdx.x;
  if (i < n) d[i] = (f16_t)s[i];
}

// dst[n*K + k] = (f16) src[k*N + n]
__global__ __launch_bounds__(256) void k_transpose_f16(const float* __restrict__ s,
                                                       f16_t* __restrict__ d,
                                                       int K, int N) {
  int i = blockIdx.x * 256 + threadIdx.x;
  if (i < K * N) {
    int n = i / K;
    int k = i - n * K;
    d[i] = (f16_t)s[(size_t)k * N + n];
  }
}

// ---- GEMM1: qkv = x @ W_attn + b_attn --------------------------------------
// q,k columns (col < 2048) -> qkv[row][col] f16 (row-major, ld 3072)
// v columns  (col >= 2048) -> vt[(b*16+h)*64 + d][t] f16 (transposed, ld 2048)

__global__ __launch_bounds__(256) void k_gemm_qkv(const f16_t* __restrict__ xh,
                                                  const f16_t* __restrict__ wat,
                                                  const float* __restrict__ bias,
                                                  f16_t* __restrict__ qkv,
                                                  f16_t* __restrict__ vt) {
  const int wave = blockIdx.x * 8 + (threadIdx.x >> 5);
  const int lane = threadIdx.x & 31;
  const int r = lane & 15, hf = lane >> 4;
  const int mt = wave / 48, nt = wave - mt * 48;   // 128 x 48 tiles of 64x64
  const int r0 = mt * 64, c0 = nt * 64;

  v8f acc[4][4];
#pragma unroll
  for (int i = 0; i < 4; ++i)
#pragma unroll
    for (int j = 0; j < 4; ++j) acc[i][j] = vzero8();

  for (int kk = 0; kk < NEMBD; kk += 32) {
    v16h A[4], Bf[4];
#pragma unroll
    for (int i = 0; i < 4; ++i) {
      const f16_t* p = xh + (size_t)(r0 + i * 16 + r) * NEMBD + kk + hf * 16;
      __builtin_prefetch(p + 32, 0, 3);   // WGP-scope prefetch of next K-slab
      A[i] = ld_a32(p);
    }
#pragma unroll
    for (int j = 0; j < 4; ++j) {
      const f16_t* p = wat + (size_t)(c0 + j * 16 + r) * NEMBD + kk + hf * 16;
      __builtin_prefetch(p + 32, 0, 3);
      Bf[j] = ld_b32(p);
    }
#pragma unroll
    for (int i = 0; i < 4; ++i)
#pragma unroll
      for (int j = 0; j < 4; ++j) acc[i][j] = wmma_f16(A[i], Bf[j], acc[i][j]);
  }

  const int bb = r0 >> 11;          // batch (tiles never straddle batches)
#pragma unroll
  for (int j = 0; j < 4; ++j) {
    const int col = c0 + j * 16 + r;
    const float bv = bias[col];
#pragma unroll
    for (int i = 0; i < 4; ++i) {
#pragma unroll
      for (int rr = 0; rr < 8; ++rr) {
        const int m = hf * 8 + rr;            // C-frag row within tile
        const int row = r0 + i * 16 + m;
        const float val = acc[i][j][rr] + bv;
        if (col < 2 * NEMBD) {
          qkv[(size_t)row * N3C + col] = (f16_t)val;
        } else {
          const int c2 = col - 2 * NEMBD;
          const int hh = c2 >> 6, dd = c2 & 63;
          const int t = row & (TSEQ - 1);
          vt[(size_t)((bb * NHEAD + hh) * HDIM + dd) * TSEQ + t] = (f16_t)val;
        }
      }
    }
  }
}

// ---- Flash attention --------------------------------------------------------
// One block per (b, h, 128 query rows); 8 waves x 16 rows.  K/V tiles of 32
// keys are staged in LDS by the Tensor Data Mover (wave 0), double-buffered,
// synchronized with s_wait_tensorcnt + block barriers.  All waves run the
// block-uniform iteration count; early-finished waves see fully-masked tiles
// (alpha = 1, contribution 0), keeping barriers uniform.

__global__ __launch_bounds__(256) void k_attn(const f16_t* __restrict__ qkv,
                                              const f16_t* __restrict__ vt,
                                              f16_t* __restrict__ yh) {
  __shared__ __align__(16) f16_t Kt[2][32][64];   // [key][d]
  __shared__ __align__(16) f16_t Vt[2][64][32];   // [d][t]
  __shared__ __align__(16) f16_t pbuf[8][16 * 32];

  const int wid = threadIdx.x >> 5;
  const int lane = threadIdx.x & 31;
  const int r = lane & 15, hf = lane >> 4;
  const int blk = blockIdx.x;
  const int b = blk >> 8;
  const int h = (blk >> 4) & 15;
  const int qb = blk & 15;
  const int q0 = qb * 128 + wid * 16;

  // Q fragments (A-layout), d = 0..31 and 32..63
  const f16_t* qrow = qkv + (size_t)(b * TSEQ + q0 + r) * N3C + h * HDIM;
  const v16h Qa = ld_a32(qrow + hf * 16);
  const v16h Qb = ld_a32(qrow + 32 + hf * 16);

  v8f o[4];
#pragma unroll
  for (int j = 0; j < 4; ++j) o[j] = vzero8();
  float rmax[8], rsum[8];
#pragma unroll
  for (int i = 0; i < 8; ++i) { rmax[i] = -1.0e30f; rsum[i] = 0.0f; }

  f16_t* pb = &pbuf[wid][0];

  const f16_t* kg = qkv + (size_t)(b * TSEQ) * N3C + NEMBD + h * HDIM;  // +key*3072
  const f16_t* vg = vt + (size_t)((b * NHEAD + h) * HDIM) * TSEQ;       // +d*2048+t
  const int niter = 4 * (qb + 1);   // 32-key tiles covering keys 0 .. qb*128+127

  if (wid == 0) {
    tdm_load_2d((unsigned int)(uintptr_t)&Kt[0][0][0], kg,
                HDIM, TSEQ, HDIM, 32, N3C);
    tdm_load_2d((unsigned int)(uintptr_t)&Vt[0][0][0], vg,
                TSEQ, HDIM, 32, HDIM, TSEQ);
  }

  for (int it = 0; it < niter; ++it) {
    const int j0 = it * 32;
    const int cb = it & 1;
    if (wid == 0) __builtin_amdgcn_s_wait_tensorcnt(0);  // tile it resident
    __syncthreads();
    if (wid == 0 && it + 1 < niter) {                    // prefetch tile it+1
      const int jn = j0 + 32;
      tdm_load_2d((unsigned int)(uintptr_t)&Kt[(it + 1) & 1][0][0],
                  kg + (size_t)jn * N3C, HDIM, TSEQ, HDIM, 32, N3C);
      tdm_load_2d((unsigned int)(uintptr_t)&Vt[(it + 1) & 1][0][0],
                  vg + jn, TSEQ, HDIM, 32, HDIM, TSEQ);
    }

    const f16_t* kt = &Kt[cb][0][0];
    const v16h K0a = ld_b32(kt + (size_t)r * 64 + hf * 16);
    const v16h K0b = ld_b32(kt + (size_t)r * 64 + 32 + hf * 16);
    const v16h K1a = ld_b32(kt + (size_t)(16 + r) * 64 + hf * 16);
    const v16h K1b = ld_b32(kt + (size_t)(16 + r) * 64 + 32 + hf * 16);

    v8f Sa = vzero8(), Sb = vzero8();
    Sa = wmma_f16(Qa, K0a, Sa); Sa = wmma_f16(Qb, K0b, Sa);
    Sb = wmma_f16(Qa, K1a, Sb); Sb = wmma_f16(Qb, K1b, Sb);

    float tm[8];
#pragma unroll
    for (int rr = 0; rr < 8; ++rr) {
      const int m = hf * 8 + rr;                 // query row in tile
      float sa = Sa[rr] * 0.125f;                // 1/sqrt(64)
      float sb = Sb[rr] * 0.125f;
      if (j0 + r      > q0 + m) sa = -1.0e30f;   // causal mask
      if (j0 + 16 + r > q0 + m) sb = -1.0e30f;
      Sa[rr] = sa; Sb[rr] = sb;
      tm[rr] = fmaxf(sa, sb);
    }
#pragma unroll
    for (int rr = 0; rr < 8; ++rr) {             // 16-lane row reductions
      tm[rr] = fmaxf(tm[rr], __shfl_xor(tm[rr], 1, 32));
      tm[rr] = fmaxf(tm[rr], __shfl_xor(tm[rr], 2, 32));
      tm[rr] = fmaxf(tm[rr], __shfl_xor(tm[rr], 4, 32));
      tm[rr] = fmaxf(tm[rr], __shfl_xor(tm[rr], 8, 32));
    }
    float alpha[8], ts[8];
#pragma unroll
    for (int rr = 0; rr < 8; ++rr) {
      const float nm = fmaxf(rmax[rr], tm[rr]);
      alpha[rr] = __expf(rmax[rr] - nm);
      rmax[rr] = nm;
      const float pa = __expf(Sa[rr] - nm);
      const float pv = __expf(Sb[rr] - nm);
      Sa[rr] = pa; Sb[rr] = pv;
      ts[rr] = pa + pv;
    }
#pragma unroll
    for (int rr = 0; rr < 8; ++rr) {
      ts[rr] += __shfl_xor(ts[rr], 1, 32);
      ts[rr] += __shfl_xor(ts[rr], 2, 32);
      ts[rr] += __shfl_xor(ts[rr], 4, 32);
      ts[rr] += __shfl_xor(ts[rr], 8, 32);
      rsum[rr] = rsum[rr] * alpha[rr] + ts[rr];
    }
#pragma unroll
    for (int j = 0; j < 4; ++j)
#pragma unroll
      for (int rr = 0; rr < 8; ++rr) o[j][rr] *= alpha[rr];

    // C-layout (col per lane) -> A-layout (row per lane) via per-wave LDS
#pragma unroll
    for (int rr = 0; rr < 8; ++rr) {
      const int m = hf * 8 + rr;
      pb[m * 32 + r]      = (f16_t)Sa[rr];
      pb[m * 32 + 16 + r] = (f16_t)Sb[rr];
    }
    const v16h P = ld_a32(pb + r * 32 + hf * 16);   // per-wave LDS is in-order

    const f16_t* vtp = &Vt[cb][0][0];
#pragma unroll
    for (int j = 0; j < 4; ++j) {
      const v16h V = ld_b32(vtp + (size_t)(j * 16 + r) * 32 + hf * 16);
      o[j] = wmma_f16(P, V, o[j]);
    }
    __syncthreads();   // all done reading buffer cb before it is overwritten
  }

#pragma unroll
  for (int j = 0; j < 4; ++j) {
#pragma unroll
    for (int rr = 0; rr < 8; ++rr) {
      const int row = b * TSEQ + q0 + hf * 8 + rr;
      const int col = h * HDIM + j * 16 + r;
      yh[(size_t)row * NEMBD + col] = (f16_t)(o[j][rr] / rsum[rr]);
    }
  }
}

// ---- GEMM2: out = y @ W_proj + b_proj (f32 output) -------------------------

__global__ __launch_bounds__(256) void k_gemm_out(const f16_t* __restrict__ yh,
                                                  const f16_t* __restrict__ wpt,
                                                  const float* __restrict__ bias,
                                                  float* __restrict__ out) {
  const int wave = blockIdx.x * 8 + (threadIdx.x >> 5);
  const int lane = threadIdx.x & 31;
  const int r = lane & 15, hf = lane >> 4;
  const int mt = wave >> 4, nt = wave & 15;       // 128 x 16 tiles of 64x64
  const int r0 = mt * 64, c0 = nt * 64;

  v8f acc[4][4];
#pragma unroll
  for (int i = 0; i < 4; ++i)
#pragma unroll
    for (int j = 0; j < 4; ++j) acc[i][j] = vzero8();

  for (int kk = 0; kk < NEMBD; kk += 32) {
    v16h A[4], Bf[4];
#pragma unroll
    for (int i = 0; i < 4; ++i) {
      const f16_t* p = yh + (size_t)(r0 + i * 16 + r) * NEMBD + kk + hf * 16;
      __builtin_prefetch(p + 32, 0, 3);
      A[i] = ld_a32(p);
    }
#pragma unroll
    for (int j = 0; j < 4; ++j) {
      const f16_t* p = wpt + (size_t)(c0 + j * 16 + r) * NEMBD + kk + hf * 16;
      __builtin_prefetch(p + 32, 0, 3);
      Bf[j] = ld_b32(p);
    }
#pragma unroll
    for (int i = 0; i < 4; ++i)
#pragma unroll
      for (int j = 0; j < 4; ++j) acc[i][j] = wmma_f16(A[i], Bf[j], acc[i][j]);
  }

#pragma unroll
  for (int j = 0; j < 4; ++j) {
    const int col = c0 + j * 16 + r;
    const float bv = bias[col];
#pragma unroll
    for (int i = 0; i < 4; ++i) {
#pragma unroll
      for (int rr = 0; rr < 8; ++rr) {
        const int row = r0 + i * 16 + hf * 8 + rr;
        out[(size_t)row * NEMBD + col] = acc[i][j][rr] + bv;
      }
    }
  }
}

// ---- Launch -----------------------------------------------------------------

extern "C" void kernel_launch(void* const* d_in, const int* in_sizes, int n_in,
                              void* d_out, int out_size, void* d_ws, size_t ws_size,
                              hipStream_t stream) {
  const float* x      = (const float*)d_in[0];   // [4,2048,1024]
  const float* W_attn = (const float*)d_in[1];   // [1024,3072]
  const float* b_attn = (const float*)d_in[2];   // [3072]
  const float* W_proj = (const float*)d_in[3];   // [1024,1024]
  const float* b_proj = (const float*)d_in[4];   // [1024]
  float* out = (float*)d_out;                    // [4,2048,1024] f32

  char* ws = (char*)d_ws;
  f16_t* xh  = (f16_t*)(ws + 0);                  //  8,388,608 f16 = 16 MB
  f16_t* wat = (f16_t*)(ws + 16777216);           //  3,145,728 f16 =  6 MB
  f16_t* wpt = (f16_t*)(ws + 23068672);           //  1,048,576 f16 =  2 MB
  f16_t* qkv = (f16_t*)(ws + 25165824);           // 25,165,824 f16 = 48 MB (q,k used)
  f16_t* vt  = (f16_t*)(ws + 75497472);           //  8,388,608 f16 = 16 MB
  f16_t* yh  = (f16_t*)(ws + 92274688);           //  8,388,608 f16 = 16 MB

  const int nx = ROWS * NEMBD;                    // 8,388,608
  k_f32_to_f16<<<nx / 256, 256, 0, stream>>>(x, xh, nx);
  k_transpose_f16<<<(N3C * NEMBD) / 256, 256, 0, stream>>>(W_attn, wat, NEMBD, N3C);
  k_transpose_f16<<<(NEMBD * NEMBD) / 256, 256, 0, stream>>>(W_proj, wpt, NEMBD, NEMBD);

  k_gemm_qkv<<<(128 * 48) / 8, 256, 0, stream>>>(xh, wat, b_attn, qkv, vt);
  k_attn<<<1024, 256, 0, stream>>>(qkv, vt, yh);   // (b,h,128-row) blocks
  k_gemm_out<<<(128 * 16) / 8, 256, 0, stream>>>(yh, wpt, b_proj, out);
}
